// LlamaModel_60962765799570
// MI455X (gfx1250) — matrix operand order repaired and checked by reading, not compiled, not run on hardware
//
#include <hip/hip_runtime.h>

// ---------------- types ----------------
typedef __bf16 v2bf  __attribute__((ext_vector_type(2)));
typedef __bf16 v8bf  __attribute__((ext_vector_type(8)));
typedef __bf16 v16bf __attribute__((ext_vector_type(16)));
typedef float  v8f   __attribute__((ext_vector_type(8)));

#define WMMA_BF16(a, b, c) \
  __builtin_amdgcn_wmma_f32_16x16x32_bf16(false, (a), false, (b), (short)0, (c), false, false)

// A-fragment (16x32 bf16, M=lane&15): lane holds contiguous K
// [16*half, 16*half+16). rowptr points at K=0 of the
// lane's row (contiguous 32 K elements).
__device__ __forceinline__ v16bf load_a_frag(const __bf16* rowptr, int half) {
  v8bf lo = *(const v8bf*)(rowptr + 16 * half);
  v8bf hi = *(const v8bf*)(rowptr + 16 * half + 8);
  return __builtin_shufflevector(lo, hi, 0, 1, 2, 3, 4, 5, 6, 7,
                                 8, 9, 10, 11, 12, 13, 14, 15);
}

__device__ __forceinline__ v8f vzero8() {
  v8f z = {0.f, 0.f, 0.f, 0.f, 0.f, 0.f, 0.f, 0.f};
  return z;
}

// ---------------- embedding gather ----------------
__global__ void embed_kernel(const int* __restrict__ toks,
                             const float* __restrict__ emb,
                             float* __restrict__ x) {
  int i = blockIdx.x * blockDim.x + threadIdx.x;   // T*D threads
  int t = i >> 10;
  int d = i & 1023;
  x[i] = emb[(size_t)toks[t] * 1024 + d];
}

// ---------------- rmsnorm: fp32 in, bf16 out ----------------
__global__ void __launch_bounds__(256)
rmsnorm_kernel(const float* __restrict__ x, const float* __restrict__ w,
               __bf16* __restrict__ out) {
  __shared__ float red[8];
  __shared__ float s_inv;
  const int t = blockIdx.x;
  const float* xr = x + (size_t)t * 1024;
  float s = 0.f;
  for (int d = threadIdx.x; d < 1024; d += 256) {
    float v = xr[d];
    s += v * v;
  }
  for (int o = 16; o > 0; o >>= 1) s += __shfl_down(s, o, 32);
  if ((threadIdx.x & 31) == 0) red[threadIdx.x >> 5] = s;
  __syncthreads();
  if (threadIdx.x == 0) {
    float tot = 0.f;
    for (int i = 0; i < 8; ++i) tot += red[i];
    s_inv = rsqrtf(tot * (1.0f / 1024.0f) + 1e-5f);
  }
  __syncthreads();
  float inv = s_inv;
  __bf16* orow = out + (size_t)t * 1024;
  for (int d = threadIdx.x; d < 1024; d += 256)
    orow[d] = (__bf16)(xr[d] * inv * w[d]);
}

// ---------------- GEMM: C(MxN) = A_bf16(MxK) @ B_f32(KxN), f32 accum -------
// 256 threads / 8 waves, 128x128 tile, BK=32. Waves arranged 4(M) x 2(N);
// each wave owns 32x64 -> 8 WMMA accumulators. A tile is staged with
// GLOBAL_LOAD_ASYNC_TO_LDS (ASYNCcnt), B tile is converted fp32->bf16 and
// stored K-major as packed dwords.
template <bool RESADD>
__global__ void __launch_bounds__(256)
gemm_kernel(const __bf16* __restrict__ A, const float* __restrict__ B,
            float* __restrict__ C, int M, int N, int K) {
  __shared__ __bf16 As[128 * 48];  // row-major [m][k], stride 48
  __shared__ __bf16 Bs[128 * 48];  // K-contiguous per column: [n][k], stride 48
  const int tid = threadIdx.x;
  const int lane = tid & 31;
  const int wave = tid >> 5;
  const int wm = wave >> 1;   // 0..3
  const int wn = wave & 1;    // 0..1
  const int lh = lane >> 4;   // half
  const int ll = lane & 15;
  const int bm = blockIdx.y * 128;
  const int bn = blockIdx.x * 128;

  v8f acc[2][4];
#pragma unroll
  for (int mi = 0; mi < 2; ++mi)
#pragma unroll
    for (int ni = 0; ni < 4; ++ni) acc[mi][ni] = vzero8();

  // A staging: 2 threads per row, 16 bf16 (32B) each via 2 async b128.
  const int arow = tid >> 1, ah = (tid & 1) * 16;
  // B staging: thread owns k rows {2*bkp, 2*bkp+1} x 8 columns.
  const int bkp = tid >> 4;        // 0..15
  const int bng = (tid & 15) * 8;  // 0..120
  unsigned* Bs32 = (unsigned*)Bs;
  // LDS byte offset of this thread's A destination (generic ptr low 32 bits
  // are the LDS offset inside the shared aperture).
  const unsigned a_lds = (unsigned)(size_t)(&As[arow * 48 + ah]);

  for (int kt = 0; kt < K; kt += 32) {
    // ---- A tile: async global -> LDS, tracked by ASYNCcnt ----
    const __bf16* ag = A + (size_t)(bm + arow) * K + kt + ah;
    asm volatile(
        "global_load_async_to_lds_b128 %0, %1, off\n\t"
        "global_load_async_to_lds_b128 %0, %1, off offset:16"
        :: "v"(a_lds), "v"(ag)
        : "memory");

    // ---- B tile: coalesced fp32 rows, convert, packed K-major stores ----
    const float* bg0 = B + (size_t)(kt + 2 * bkp) * N + bn + bng;
    const float* bg1 = bg0 + N;
    float4 r0a = *(const float4*)(bg0);
    float4 r0b = *(const float4*)(bg0 + 4);
    float4 r1a = *(const float4*)(bg1);
    float4 r1b = *(const float4*)(bg1 + 4);
    if (kt + 32 < K) {  // hint next K-tile of the weight stream (L2 resident)
      __builtin_prefetch(bg0 + (size_t)32 * N, 0, 1);
      __builtin_prefetch(bg1 + (size_t)32 * N, 0, 1);
    }
    float r0[8] = {r0a.x, r0a.y, r0a.z, r0a.w, r0b.x, r0b.y, r0b.z, r0b.w};
    float r1[8] = {r1a.x, r1a.y, r1a.z, r1a.w, r1b.x, r1b.y, r1b.z, r1b.w};
#pragma unroll
    for (int i = 0; i < 8; ++i) {
      v2bf p = {(__bf16)r0[i], (__bf16)r1[i]};  // (k=2bkp, k=2bkp+1) adjacent
      Bs32[(bng + i) * 24 + bkp] = __builtin_bit_cast(unsigned, p);
    }

    asm volatile("s_wait_asynccnt 0" ::: "memory");
    __syncthreads();

    v16bf af[2], bfm[4];
#pragma unroll
    for (int mi = 0; mi < 2; ++mi)
      af[mi] = load_a_frag(&As[(wm * 32 + mi * 16 + ll) * 48], lh);
#pragma unroll
    for (int ni = 0; ni < 4; ++ni)
      bfm[ni] = *(const v16bf*)&Bs[(wn * 64 + ni * 16 + ll) * 48 + 16 * lh];
#pragma unroll
    for (int mi = 0; mi < 2; ++mi)
#pragma unroll
      for (int ni = 0; ni < 4; ++ni)
        acc[mi][ni] = WMMA_BF16(af[mi], bfm[ni], acc[mi][ni]);
    __syncthreads();
  }

  // store: C/D layout -> element (m = r + 8*half, n = lane&15)
#pragma unroll
  for (int mi = 0; mi < 2; ++mi)
#pragma unroll
    for (int ni = 0; ni < 4; ++ni) {
      const int col = bn + wn * 64 + ni * 16 + ll;
#pragma unroll
      for (int r = 0; r < 8; ++r) {
        const int row = bm + wm * 32 + mi * 16 + 8 * lh + r;
        const size_t idx = (size_t)row * N + col;
        float v = acc[mi][ni][r];
        if (RESADD)
          C[idx] += v;
        else
          C[idx] = v;
      }
    }
}

// ---------------- RoPE (fp32 in, bf16 out, layout [t][head*64+d]) ----------
__global__ void rope_kernel(const float* __restrict__ in,
                            __bf16* __restrict__ out, int nh, int n) {
  int idx = blockIdx.x * blockDim.x + threadIdx.x;  // T*nh*32
  if (idx >= n) return;
  int i2 = idx & 31;
  int h = (idx >> 5) % nh;
  int t = idx / (nh * 32);
  float inv = __powf(10000.0f, -(float)(2 * i2) / 64.0f);
  float ang = (float)t * inv;
  float sv, cv;
  __sincosf(ang, &sv, &cv);
  int base = (t * nh + h) * 64 + 2 * i2;
  float x0 = in[base], x1 = in[base + 1];
  out[base] = (__bf16)(x0 * cv - x1 * sv);
  out[base + 1] = (__bf16)(x0 * sv + x1 * cv);
}

// ---------------- V transpose: [t][kv*64+d] f32 -> [kv][d][t] bf16 ---------
__global__ void transpose_v_kernel(const float* __restrict__ vf,
                                   __bf16* __restrict__ vt) {
  int idx = blockIdx.x * blockDim.x + threadIdx.x;  // T*256
  int d = idx & 63;
  int kv = (idx >> 6) & 3;
  int t = idx >> 8;
  vt[((size_t)kv * 64 + d) * 2048 + t] = (__bf16)vf[idx];
}

// ---------------- flash attention, 1 wave per (head, 16-query block) -------
__global__ void __launch_bounds__(32)
attn_kernel(const __bf16* __restrict__ qbf, const __bf16* __restrict__ kbf,
            const __bf16* __restrict__ vtbf, __bf16* __restrict__ obf) {
  const int h = blockIdx.x;   // 0..15
  const int qb = blockIdx.y;  // 0..127
  const int lane = threadIdx.x;
  const int lh = lane >> 4;
  const int ll = lane & 15;
  const int kv = h >> 2;  // GQA: 4 query heads per kv head

  __shared__ float S_lds[16 * 32];
  __shared__ __bf16 P_lds[16 * 32];
  __shared__ float alpha_lds[16];
  __shared__ float scl_lds[16];

  v8f O0 = vzero8(), O1 = vzero8(), O2 = vzero8(), O3 = vzero8();
  float m_run = -__builtin_inff();
  float l_run = 0.f;
  const int tq = qb * 16 + lane;  // meaningful for lane < 16

  // Q A-fragments, constant across key blocks (d 0..31 and 32..63)
  const __bf16* qrow = qbf + (size_t)(qb * 16 + ll) * 1024 + h * 64;
  v16bf qf0 = load_a_frag(qrow, lh);
  v16bf qf1 = load_a_frag(qrow + 32, lh);

  const int kb_end = (qb * 16 + 15) >> 5;
  for (int kb = 0; kb <= kb_end; ++kb) {
    const int k0 = kb * 32;
    v8f s0 = vzero8(), s1 = vzero8();
    {
      // B[d][key] = K[key][d]; kbf layout [t][kv*64+d], contiguous d
      const __bf16* kbase = kbf + kv * 64 + 16 * lh;
      v16bf b00 = *(const v16bf*)(kbase + (size_t)(k0 + ll) * 256);
      v16bf b01 = *(const v16bf*)(kbase + (size_t)(k0 + 16 + ll) * 256);
      s0 = WMMA_BF16(qf0, b00, s0);
      s1 = WMMA_BF16(qf0, b01, s1);
      v16bf b10 = *(const v16bf*)(kbase + 32 + (size_t)(k0 + ll) * 256);
      v16bf b11 = *(const v16bf*)(kbase + 32 + (size_t)(k0 + 16 + ll) * 256);
      s0 = WMMA_BF16(qf1, b10, s0);
      s1 = WMMA_BF16(qf1, b11, s1);
    }
#pragma unroll
    for (int r = 0; r < 8; ++r) {
      int m = r + 8 * lh;
      S_lds[m * 32 + ll] = s0[r];
      S_lds[m * 32 + 16 + ll] = s1[r];
    }
    __syncthreads();
    if (lane < 16) {  // one row per lane: online softmax
      float vals[32];
      float mx = m_run;
#pragma unroll
      for (int n = 0; n < 32; ++n) {
        float sv = S_lds[lane * 32 + n] * 0.125f;  // 1/sqrt(64)
        if (k0 + n > tq) sv = -__builtin_inff();   // causal mask
        vals[n] = sv;
        mx = fmaxf(mx, sv);
      }
      float alpha = __expf(m_run - mx);
      float sum = 0.f;
#pragma unroll
      for (int n = 0; n < 32; ++n) {
        float p = __expf(vals[n] - mx);
        sum += p;
        P_lds[lane * 32 + n] = (__bf16)p;
      }
      l_run = l_run * alpha + sum;
      m_run = mx;
      alpha_lds[lane] = alpha;
    }
    __syncthreads();
#pragma unroll
    for (int r = 0; r < 8; ++r) {
      float a = alpha_lds[r + 8 * lh];
      O0[r] *= a; O1[r] *= a; O2[r] *= a; O3[r] *= a;
    }
    v16bf pf = load_a_frag(&P_lds[ll * 32], lh);
    // B[key][d] = vtbf[kv][d][t], contiguous t
    const __bf16* vbase = vtbf + (size_t)kv * 64 * 2048 + k0 + 16 * lh;
    v16bf v0 = *(const v16bf*)(vbase + (size_t)(ll + 0) * 2048);
    v16bf v1 = *(const v16bf*)(vbase + (size_t)(ll + 16) * 2048);
    v16bf v2 = *(const v16bf*)(vbase + (size_t)(ll + 32) * 2048);
    v16bf v3 = *(const v16bf*)(vbase + (size_t)(ll + 48) * 2048);
    O0 = WMMA_BF16(pf, v0, O0);
    O1 = WMMA_BF16(pf, v1, O1);
    O2 = WMMA_BF16(pf, v2, O2);
    O3 = WMMA_BF16(pf, v3, O3);
    __syncthreads();
  }
  if (lane < 16) scl_lds[lane] = 1.0f / l_run;
  __syncthreads();
#pragma unroll
  for (int r = 0; r < 8; ++r) {
    float inv = scl_lds[r + 8 * lh];
    int t = qb * 16 + r + 8 * lh;
    __bf16* op = obf + (size_t)t * 1024 + h * 64 + ll;
    op[0] = (__bf16)(O0[r] * inv);
    op[16] = (__bf16)(O1[r] * inv);
    op[32] = (__bf16)(O2[r] * inv);
    op[48] = (__bf16)(O3[r] * inv);
  }
}

// ---------------- silu(gate) * up -> bf16 ----------------
__global__ void silu_mul_kernel(const float* __restrict__ g,
                                const float* __restrict__ u,
                                __bf16* __restrict__ o, int n) {
  int i = blockIdx.x * blockDim.x + threadIdx.x;
  if (i >= n) return;
  float x = g[i];
  float s = x / (1.0f + __expf(-x));
  o[i] = (__bf16)(s * u[i]);
}

// ---------------- host orchestration ----------------
extern "C" void kernel_launch(void* const* d_in, const int* in_sizes, int n_in,
                              void* d_out, int out_size, void* d_ws,
                              size_t ws_size, hipStream_t stream) {
  (void)in_sizes; (void)n_in; (void)out_size; (void)ws_size;
  constexpr int T = 2048, D = 1024, II = 2816, V = 32000;

  const int* toks = (const int*)d_in[0];
  const float* emb = (const float*)d_in[1];
  const float* Wq = (const float*)d_in[2];
  const float* Wk = (const float*)d_in[3];
  const float* Wv = (const float*)d_in[4];
  const float* Wo = (const float*)d_in[5];
  const float* Wg = (const float*)d_in[6];
  const float* Wu = (const float*)d_in[7];
  const float* Wd = (const float*)d_in[8];
  const float* ln1 = (const float*)d_in[9];
  const float* ln2 = (const float*)d_in[10];
  const float* nw = (const float*)d_in[11];
  const float* Wlm = (const float*)d_in[12];
  float* out = (float*)d_out;

  char* ws = (char*)d_ws;
  size_t off = 0;
  auto alloc = [&](size_t bytes) {
    char* p = ws + off;
    off += (bytes + 255) & ~(size_t)255;
    return (void*)p;
  };
  float* x = (float*)alloc((size_t)T * D * 4);
  __bf16* hbf = (__bf16*)alloc((size_t)T * D * 2);
  float* qf = (float*)alloc((size_t)T * 1024 * 4);
  float* kf = (float*)alloc((size_t)T * 256 * 4);
  float* vf = (float*)alloc((size_t)T * 256 * 4);
  __bf16* qbf = (__bf16*)alloc((size_t)T * 1024 * 2);
  __bf16* kbf = (__bf16*)alloc((size_t)T * 256 * 2);
  __bf16* vtbf = (__bf16*)alloc((size_t)4 * 64 * T * 2);
  __bf16* abf = (__bf16*)alloc((size_t)T * 1024 * 2);
  float* gatef = (float*)alloc((size_t)T * II * 4);
  float* upf = (float*)alloc((size_t)T * II * 4);
  __bf16* actbf = (__bf16*)alloc((size_t)T * II * 2);

  embed_kernel<<<T * D / 256, 256, 0, stream>>>(toks, emb, x);
  for (int l = 0; l < 2; ++l) {
    rmsnorm_kernel<<<T, 256, 0, stream>>>(x, ln1 + (size_t)l * D, hbf);
    gemm_kernel<false><<<dim3(1024 / 128, T / 128), 256, 0, stream>>>(
        hbf, Wq + (size_t)l * D * 1024, qf, T, 1024, D);
    gemm_kernel<false><<<dim3(256 / 128, T / 128), 256, 0, stream>>>(
        hbf, Wk + (size_t)l * D * 256, kf, T, 256, D);
    gemm_kernel<false><<<dim3(256 / 128, T / 128), 256, 0, stream>>>(
        hbf, Wv + (size_t)l * D * 256, vf, T, 256, D);
    rope_kernel<<<T * 16 * 32 / 256, 256, 0, stream>>>(qf, qbf, 16, T * 16 * 32);
    rope_kernel<<<T * 4 * 32 / 256, 256, 0, stream>>>(kf, kbf, 4, T * 4 * 32);
    transpose_v_kernel<<<T * 256 / 256, 256, 0, stream>>>(vf, vtbf);
    attn_kernel<<<dim3(16, T / 16), 32, 0, stream>>>(qbf, kbf, vtbf, abf);
    gemm_kernel<true><<<dim3(1024 / 128, T / 128), 256, 0, stream>>>(
        abf, Wo + (size_t)l * 1024 * 1024, x, T, 1024, 1024);
    rmsnorm_kernel<<<T, 256, 0, stream>>>(x, ln2 + (size_t)l * D, hbf);
    gemm_kernel<false><<<dim3(II / 128, T / 128), 256, 0, stream>>>(
        hbf, Wg + (size_t)l * D * II, gatef, T, II, D);
    gemm_kernel<false><<<dim3(II / 128, T / 128), 256, 0, stream>>>(
        hbf, Wu + (size_t)l * D * II, upf, T, II, D);
    silu_mul_kernel<<<T * II / 256, 256, 0, stream>>>(gatef, upf, actbf, T * II);
    gemm_kernel<true><<<dim3(1024 / 128, T / 128), 256, 0, stream>>>(
        actbf, Wd + (size_t)l * II * 1024, x, T, 1024, II);
  }
  rmsnorm_kernel<<<T, 256, 0, stream>>>(x, nw, hbf);
  gemm_kernel<false><<<dim3(V / 128, T / 128), 256, 0, stream>>>(hbf, Wlm, out,
                                                                 T, V, D);
}
